// NeuronMLPBlock_72438918414393
// MI455X (gfx1250) — matrix operand order repaired and checked by reading, not compiled, not run on hardware
//
#include <hip/hip_runtime.h>
#include <hip/hip_bf16.h>

typedef unsigned int  uint;
typedef unsigned short ushort;
typedef unsigned long long u64;

typedef __attribute__((ext_vector_type(16))) __bf16 v16bf;
typedef __attribute__((ext_vector_type(8)))  float  v8f;

#define Hdim 2048
#define Idim 4096
#define Tn   2048      // tokens = B*S
#define En   8
#define ICn  1024      // I-chunk per LDS pass
#define MR   32        // token rows per block (2 WMMA M-subtiles)
#define MTn  64        // Tn/MR max M-tiles per expert

// ---------- helpers ----------------------------------------------------------

// Truncating f32->bf16 pack of two floats in ONE VALU op (v_perm_b32):
// result = hi16(hi)<<16 | hi16(lo)
__device__ __forceinline__ uint pack2bf(float lo, float hi) {
  union { float f; uint u; } a, b;
  a.f = hi; b.f = lo;
  return __builtin_amdgcn_perm(a.u, b.u, 0x07060302u);
}
__device__ __forceinline__ ushort bf16trunc(float f) {
  union { float f; uint u; } c; c.f = f;
  return (ushort)(c.u >> 16);
}

// A fragment (16x32 bf16) from row-major bf16 LDS tile, CDNA5 16-bit A layout:
// lanes 0-15 (g=0): v0-3 K=0..7, v4-7 K=16..23 ; lanes 16-31 (g=1): +8
__device__ __forceinline__ v16bf load_a_frag_lds(const uint* base_u, int row_stride_u,
                                                 int mbase, int k0, int lane) {
  int m = (lane & 15) + mbase, g = lane >> 4;
  const uint* p = base_u + m * row_stride_u + (k0 >> 1) + g * 4;
  union { uint4 q[2]; v16bf v; } u;
  u.q[0] = *(const uint4*)(p);       // ds_load_b128
  u.q[1] = *(const uint4*)(p + 8);   // ds_load_b128
  return u.v;
}

// B fragment (32x16 bf16) from global fp32 W, element (k,n) at W[k*ldw + n].
// lane: n = n0 + (lane&15); K half g = lane>>4; VGPR v<4 holds K = 8g+2v, +1; v>=4 holds K = 16+8g+2(v-4), +1
__device__ __forceinline__ v16bf load_b_frag_g(const float* __restrict__ W, int ldw,
                                               int k0, int n0, int lane) {
  int n = n0 + (lane & 15);
  const float* p = W + (size_t)(k0 + ((lane >> 4) << 3)) * ldw + n;
  union { uint u[8]; v16bf v; } r;
#pragma unroll
  for (int v = 0; v < 8; ++v) {
    float f0 = p[0];
    float f1 = p[ldw];
    p += (v == 3 ? 10 : 2) * ldw;
    r.u[v] = pack2bf(f0, f1);        // one v_perm_b32
  }
  return r.v;
}

__device__ __forceinline__ v8f wmma_bf16(v16bf a, v16bf b, v8f c) {
  return __builtin_amdgcn_wmma_f32_16x16x32_bf16(false, a, false, b, (short)0, c,
                                                 false, false);
}

// ---------- kernel 1: zero output + expert counters --------------------------

__global__ void k_zero(float* __restrict__ out, int n, int* __restrict__ cnt) {
  int i = blockIdx.x * 256 + threadIdx.x;
  if (i < n) out[i] = 0.0f;
  if (blockIdx.x == 0 && threadIdx.x < En) cnt[threadIdx.x] = 0;
}

// ---------- kernel 2: RMSNorm + router (one block per token) -----------------

__global__ __launch_bounds__(256)
void k_router(const float* __restrict__ x, const float* __restrict__ nw,
              const float* __restrict__ rw, ushort* __restrict__ Xn,
              int* __restrict__ cnt, int* __restrict__ toks,
              float* __restrict__ scls) {
  const int t = blockIdx.x, tid = threadIdx.x;
  __shared__ float red[256];
  __shared__ float lred[En * 256];

  const float* xr = x + (size_t)t * Hdim;
  float xv[8]; float ss = 0.f;
#pragma unroll
  for (int j = 0; j < 8; ++j) { xv[j] = xr[tid * 8 + j]; ss += xv[j] * xv[j]; }
  red[tid] = ss; __syncthreads();
#pragma unroll
  for (int s = 128; s > 0; s >>= 1) {
    if (tid < s) red[tid] += red[tid + s];
    __syncthreads();
  }
  const float rstd = rsqrtf(red[0] * (1.0f / Hdim) + 1e-6f);

  float lp[En];
#pragma unroll
  for (int e = 0; e < En; ++e) lp[e] = 0.f;
  float tn[8];
#pragma unroll
  for (int j = 0; j < 8; ++j) {
    int h = tid * 8 + j;
    tn[j] = xv[j] * rstd * nw[h];
#pragma unroll
    for (int e = 0; e < En; ++e) lp[e] += tn[j] * rw[h * En + e];
  }
  // store normalized token as packed bf16 (4 dwords)
  uint* Xu = (uint*)Xn;
#pragma unroll
  for (int j = 0; j < 4; ++j)
    Xu[(size_t)t * (Hdim / 2) + tid * 4 + j] = pack2bf(tn[2 * j], tn[2 * j + 1]);

#pragma unroll
  for (int e = 0; e < En; ++e) lred[e * 256 + tid] = lp[e];
  __syncthreads();
  if (tid < En) {
    float s = 0.f;
    for (int i = 0; i < 256; ++i) s += lred[tid * 256 + i];
    red[tid] = s;
  }
  __syncthreads();
  if (tid == 0) {
    float mx = red[0];
#pragma unroll
    for (int e = 1; e < En; ++e) mx = fmaxf(mx, red[e]);
    float pe[En];
#pragma unroll
    for (int e = 0; e < En; ++e) pe[e] = __expf(red[e] - mx);
    int i0 = 0;
#pragma unroll
    for (int e = 1; e < En; ++e) if (pe[e] > pe[i0]) i0 = e;
    int i1 = (i0 == 0) ? 1 : 0;
#pragma unroll
    for (int e = 0; e < En; ++e) if (e != i0 && pe[e] > pe[i1]) i1 = e;
    float v0 = pe[i0], v1 = pe[i1], inv = 1.0f / (pe[i0] + pe[i1]);
    v0 *= inv; v1 *= inv;
    int p0 = atomicAdd(&cnt[i0], 1);
    toks[i0 * Tn + p0] = t; scls[i0 * Tn + p0] = v0;
    int p1 = atomicAdd(&cnt[i1], 1);
    toks[i1 * Tn + p1] = t; scls[i1 * Tn + p1] = v1;
  }
}

// ---------- kernel 3: fused expert SwiGLU MLP (grouped-GEMM tile) ------------
// grid = En * MTn blocks, 256 threads = 8 waves. Block handles MR=32 token rows
// of one expert (2 WMMA M-subtiles so each converted B-frag feeds 2 WMMAs);
// waves split the N dimension; y accumulators (32 x v8f) stay in VGPRs.

__global__ __launch_bounds__(256)
void k_expert(const float* __restrict__ wg, const float* __restrict__ wu,
              const float* __restrict__ wd, const ushort* __restrict__ Xn,
              const int* __restrict__ cnt, const int* __restrict__ toks,
              const float* __restrict__ scls, float* __restrict__ out) {
  __shared__ __align__(16) ushort As[MR * Hdim];   // 128 KB: 32 token rows, bf16
  __shared__ __align__(16) ushort Hs[MR * ICn];    //  64 KB: silu(g)*u chunk, bf16
  __shared__ int   tok_s[MR];
  __shared__ float scl_s[MR];

  const int e  = blockIdx.x / MTn;
  const int mt = blockIdx.x % MTn;
  const int nrows = cnt[e];
  if (mt * MR >= nrows) return;

  const int tid = threadIdx.x, lane = tid & 31, wave = tid >> 5;

  if (tid < MR) {
    int r = mt * MR + tid;
    if (r < nrows) { tok_s[tid] = toks[e * Tn + r]; scl_s[tid] = scls[e * Tn + r]; }
    else           { tok_s[tid] = toks[e * Tn + mt * MR]; scl_s[tid] = 0.0f; }
  }
  __syncthreads();

  // gather MR bf16 token rows into LDS with ASYNC global->LDS b128 copies
  // (CDNA5 GLOBAL_LOAD_ASYNC_TO_LDS_B128, tracked by ASYNCcnt; no VGPR bounce)
  {
    const uint lds_base = (uint)(size_t)&As[0];    // low 32 bits = LDS offset
    for (int i = tid; i < MR * (Hdim / 8); i += 256) {   // 32 iters, EXEC all-1s
      int r = i >> 8, c = i & 255;                 // 256 uint4 per row
      u64  gaddr = (u64)(size_t)(Xn + (size_t)tok_s[r] * Hdim + c * 8);
      uint laddr = lds_base + (uint)(i * 16);
      asm volatile("global_load_async_to_lds_b128 %0, %1, off"
                   :: "v"(laddr), "v"(gaddr) : "memory");
    }
    asm volatile("s_wait_asynccnt 0x0" ::: "memory");
  }
  __syncthreads();

  const float* wge = wg + (size_t)e * Hdim * Idim;
  const float* wue = wu + (size_t)e * Hdim * Idim;
  const float* wde = wd + (size_t)e * Idim * Hdim;

  v8f yacc[32];                                    // [16 n-tiles][2 m-subtiles]
  const v8f zero8 = {0.f, 0.f, 0.f, 0.f, 0.f, 0.f, 0.f, 0.f};
#pragma unroll
  for (int i = 0; i < 32; ++i) yacc[i] = zero8;

  for (int ic = 0; ic < Idim; ic += ICn) {
    // ---- phase 1: gate/up GEMM + SiLU for I columns [ic, ic+ICn) -----------
#pragma unroll 1
    for (int nt = 0; nt < (ICn / 16) / 8; ++nt) {  // this wave: ICn/8 I-cols
      const int ncol = wave * (ICn / 8) + nt * 16; // col within chunk
      const int n0 = ic + ncol;                    // global I col
      v8f accg0 = zero8, accg1 = zero8, accu0 = zero8, accu1 = zero8;
#pragma unroll 1
      for (int k0 = 0; k0 < Hdim; k0 += 32) {
        v16bf a0 = load_a_frag_lds((const uint*)As, Hdim / 2, 0,  k0, lane);
        v16bf a1 = load_a_frag_lds((const uint*)As, Hdim / 2, 16, k0, lane);
        v16bf bg = load_b_frag_g(wge, Idim, k0, n0, lane);
        v16bf bu = load_b_frag_g(wue, Idim, k0, n0, lane);
        if (k0 + 32 < Hdim) {
          __builtin_prefetch(wge + (size_t)(k0 + 32) * Idim + n0, 0, 1);
          __builtin_prefetch(wue + (size_t)(k0 + 32) * Idim + n0, 0, 1);
        }
        accg0 = wmma_bf16(a0, bg, accg0);
        accg1 = wmma_bf16(a1, bg, accg1);
        accu0 = wmma_bf16(a0, bu, accu0);
        accu1 = wmma_bf16(a1, bu, accu1);
      }
      // h = silu(g) * u  -> bf16 into Hs (row-major [MR][ICn])
#pragma unroll
      for (int j = 0; j < 8; ++j) {
        int m   = j + ((lane >> 4) << 3);
        int col = ncol + (lane & 15);
        float g0 = accg0[j], u0 = accu0[j];
        float g1 = accg1[j], u1 = accu1[j];
        Hs[m * ICn + col]        = bf16trunc((g0 / (1.0f + __expf(-g0))) * u0);
        Hs[(m + 16) * ICn + col] = bf16trunc((g1 / (1.0f + __expf(-g1))) * u1);
      }
    }
    __syncthreads();

    // ---- phase 2: down GEMM, K over this I-chunk; wave owns 256 H-cols ----
    const float* wdc = wde + (size_t)ic * Hdim;
#pragma unroll 1
    for (int k0 = 0; k0 < ICn; k0 += 32) {
      v16bf a0 = load_a_frag_lds((const uint*)Hs, ICn / 2, 0,  k0, lane);
      v16bf a1 = load_a_frag_lds((const uint*)Hs, ICn / 2, 16, k0, lane);
#pragma unroll
      for (int nt = 0; nt < 16; ++nt) {
        v16bf bd = load_b_frag_g(wdc, Hdim, k0, wave * 256 + nt * 16, lane);
        yacc[2 * nt]     = wmma_bf16(a0, bd, yacc[2 * nt]);
        yacc[2 * nt + 1] = wmma_bf16(a1, bd, yacc[2 * nt + 1]);
      }
    }
    __syncthreads();   // before next chunk overwrites Hs
  }

  // ---- epilogue: scaled atomic scatter-add into out ------------------------
  const int mhalf = ((lane >> 4) << 3);
  const int nc    = lane & 15;
#pragma unroll
  for (int mg = 0; mg < 2; ++mg) {
#pragma unroll
    for (int j = 0; j < 8; ++j) {
      int   row = mg * 16 + mhalf + j;
      float s   = scl_s[row];
      if (s != 0.0f) {
        float* orow = out + (size_t)tok_s[row] * Hdim + wave * 256 + nc;
#pragma unroll
        for (int nt = 0; nt < 16; ++nt)
          atomicAdd(orow + nt * 16, s * yacc[2 * nt + mg][j]);
      }
    }
  }
}

// ---------- launch -----------------------------------------------------------

extern "C" void kernel_launch(void* const* d_in, const int* in_sizes, int n_in,
                              void* d_out, int out_size, void* d_ws, size_t ws_size,
                              hipStream_t stream) {
  (void)in_sizes; (void)n_in; (void)out_size; (void)ws_size;
  const float* x  = (const float*)d_in[0];
  const float* nw = (const float*)d_in[1];
  const float* rw = (const float*)d_in[2];
  const float* wg = (const float*)d_in[3];
  const float* wu = (const float*)d_in[4];
  const float* wd = (const float*)d_in[5];
  float* out = (float*)d_out;

  // workspace layout: bf16 activations | counters | token lists | scales
  ushort* Xn  = (ushort*)d_ws;
  int*    cnt = (int*)((char*)d_ws + (size_t)Tn * Hdim * 2);
  int*    tks = cnt + 64;
  float*  scl = (float*)(tks + En * Tn);

  k_zero<<<(Tn * Hdim + 255) / 256, 256, 0, stream>>>(out, Tn * Hdim, cnt);
  k_router<<<Tn, 256, 0, stream>>>(x, nw, rw, Xn, cnt, tks, scl);
  k_expert<<<En * MTn, 256, 0, stream>>>(wg, wu, wd, Xn, cnt, tks, scl, out);
}
